// MultiHeadAttention_50861002719805
// MI455X (gfx1250) — matrix operand order repaired and checked by reading, not compiled, not run on hardware
//
#include <hip/hip_runtime.h>
#include <hip/hip_bf16.h>
#include <math.h>
#include <stdint.h>

// Problem constants (match reference)
#define B_  4
#define T_  2048
#define D_  1024
#define H_  16
#define HS_ 64
#define F_  192  // 3*HS

typedef __attribute__((ext_vector_type(16))) _Float16 v16h;
typedef __attribute__((ext_vector_type(8)))  float    v8f;

__device__ __forceinline__ v8f wmma16(v16h a, v16h b, v8f c) {
  // D = A(16x32 f16) * B(32x16 f16) + C(16x16 f32)
  return __builtin_amdgcn_wmma_f32_16x16x32_f16(false, a, false, b, (short)0, c,
                                                false, false);
}

// A fragment (16x32 f16, row-major source, ld in halves):
// lane L: row = L&15; holds 16 contiguous halves [k0..k0+15], k0 = (L>>4)*16
__device__ __forceinline__ v16h load_a16(const _Float16* base, int ld) {
  const int lane = threadIdx.x & 31;
  const int row  = lane & 15;
  const int k0   = (lane >> 4) << 4;
  const _Float16* p = base + row * ld + k0;
  union { v16h v; float4 f[2]; } u;
  u.f[0] = *reinterpret_cast<const float4*>(p);
  u.f[1] = *reinterpret_cast<const float4*>(p + 8);
  return u.v;
}

// B fragment (32x16 f16). Source stored transposed as [n][k], ld in halves.
// lane L: col n = L&15; holds 16 contiguous K halves at k0 = (L>>4)*16
__device__ __forceinline__ v16h load_b16(const _Float16* baseT, int ld) {
  const int lane = threadIdx.x & 31;
  const int col  = lane & 15;
  const int k0   = (lane >> 4) << 4;
  return *reinterpret_cast<const v16h*>(baseT + col * ld + k0);
}

// C tile (16x16 f32) stores: lane L owns (row = m + 8*(L>>4), col = L&15)
__device__ __forceinline__ void store_tile_f16(_Float16* dst, int ld, const v8f& c) {
  const int lane = threadIdx.x & 31;
  const int col = lane & 15, rbase = (lane >> 4) << 3;
#pragma unroll
  for (int m = 0; m < 8; ++m) dst[(rbase + m) * ld + col] = (_Float16)c[m];
}
__device__ __forceinline__ void store_tile_f16_T(_Float16* dst, int ld, const v8f& c) {
  const int lane = threadIdx.x & 31;
  const int col = lane & 15, rbase = (lane >> 4) << 3;
#pragma unroll
  for (int m = 0; m < 8; ++m) dst[(size_t)col * ld + rbase + m] = (_Float16)c[m];
}

// ---- CDNA5 async global->LDS copy (ASYNCcnt path), per-lane 16 bytes ----
// VDST operand = LDS byte offset (u32 VGPR); addr = 64-bit global VA.
__device__ __forceinline__ void async_cp16(unsigned lds_off, const void* gptr) {
  unsigned long long ga = (unsigned long long)(uintptr_t)gptr;
  asm volatile("global_load_async_to_lds_b128 %0, %1, off"
               :: "v"(lds_off), "v"(ga) : "memory");
}

// ---------------- precision / layout prep ----------------

__global__ __launch_bounds__(256) void cast_f32_to_f16(const float* __restrict__ in,
                                                       _Float16* __restrict__ out,
                                                       int n) {
  int i = blockIdx.x * 256 + threadIdx.x;
  if (i < n) out[i] = (_Float16)in[i];
}

// W_qkv [h][c][f] -> f16 [h][f][c]  (B-operand layout for the QKV GEMM)
__global__ __launch_bounds__(256) void transpose_wqkv(const float* __restrict__ in,
                                                      _Float16* __restrict__ out) {
  int i = blockIdx.x * 256 + threadIdx.x;
  if (i >= H_ * D_ * F_) return;
  int h = i / (D_ * F_);
  int r = i - h * (D_ * F_);
  int c = r / F_;
  int f = r - c * F_;
  out[((size_t)h * F_ + f) * D_ + c] = (_Float16)in[i];
}

// W_out [c][j] -> f16 [j][c]  (B-operand layout for the output GEMM)
__global__ __launch_bounds__(256) void transpose_wout(const float* __restrict__ in,
                                                      _Float16* __restrict__ out) {
  int i = blockIdx.x * 256 + threadIdx.x;
  if (i >= D_ * D_) return;
  int c = i / D_;
  int j = i - c * D_;
  out[(size_t)j * D_ + c] = (_Float16)in[i];
}

// ---------------- QKV projection GEMM (64x64 per wave) ----------------
// f-tile of 64 is exactly one of {Q, K, V}, so the store path is uniform.

__global__ __launch_bounds__(256) void qkv_gemm(const _Float16* __restrict__ xh,
                                                const _Float16* __restrict__ wT,
                                                _Float16* __restrict__ q,
                                                _Float16* __restrict__ kk,
                                                _Float16* __restrict__ vT) {
  const int wid = blockIdx.x * 8 + (threadIdx.x >> 5);
  const int ft  = wid % 3;                   // 3 tiles of 64 over F=192
  const int tt  = (wid / 3) & 31;            // 32 tiles of 64 over T
  const int h   = (wid / 96) & 15;
  const int b   = wid / 1536;
  const int t0 = tt * 64;
  const int f0 = ft * 64;

  const _Float16* ap = xh + ((size_t)b * T_ + t0) * D_;
  const _Float16* bp = wT + ((size_t)h * F_ + f0) * D_;

  v8f acc[4][4] = {};
  for (int c0 = 0; c0 < D_; c0 += 32) {
    __builtin_prefetch(ap + c0 + 256, 0, 0);
    v16h A[4], Bv[4];
#pragma unroll
    for (int i = 0; i < 4; ++i) A[i]  = load_a16(ap + (size_t)(16 * i) * D_ + c0, D_);
#pragma unroll
    for (int j = 0; j < 4; ++j) Bv[j] = load_b16(bp + (size_t)(16 * j) * D_ + c0, D_);
#pragma unroll
    for (int i = 0; i < 4; ++i)
#pragma unroll
      for (int j = 0; j < 4; ++j) acc[i][j] = wmma16(A[i], Bv[j], acc[i][j]);
  }

  const int bh = b * H_ + h;
  if (ft == 0) {                       // Q: [bh][t][d]
    _Float16* dst = q + ((size_t)bh * T_ + t0) * HS_;
#pragma unroll
    for (int i = 0; i < 4; ++i)
#pragma unroll
      for (int j = 0; j < 4; ++j)
        store_tile_f16(dst + (size_t)(16 * i) * HS_ + 16 * j, HS_, acc[i][j]);
  } else if (ft == 1) {                // K: [bh][s][d]  (== [n][k] for QK^T)
    _Float16* dst = kk + ((size_t)bh * T_ + t0) * HS_;
#pragma unroll
    for (int i = 0; i < 4; ++i)
#pragma unroll
      for (int j = 0; j < 4; ++j)
        store_tile_f16(dst + (size_t)(16 * i) * HS_ + 16 * j, HS_, acc[i][j]);
  } else {                             // V transposed: [bh][d][s] (== [n][k] for P*V)
    _Float16* dst = vT + (size_t)bh * HS_ * T_ + t0;
#pragma unroll
    for (int i = 0; i < 4; ++i)
#pragma unroll
      for (int j = 0; j < 4; ++j)
        store_tile_f16_T(dst + (size_t)(16 * j) * T_ + 16 * i, T_, acc[i][j]);
  }
}

// ---------------- flash attention ----------------
// 128 threads = 4 waves; waves share K/V tiles staged in LDS by the async DMA
// path (double-buffered). Each wave owns a 16-row q tile.

__global__ __launch_bounds__(128) void flash_attn(const _Float16* __restrict__ q,
                                                  const _Float16* __restrict__ kk,
                                                  const _Float16* __restrict__ vT,
                                                  _Float16* __restrict__ mha) {
  const int bh = blockIdx.y;             // 0..B*H-1
  const int b = bh >> 4, h = bh & 15;
  const int w = threadIdx.x >> 5;        // wave in block, 0..3
  const int t0 = (blockIdx.x * 4 + w) * 16;
  const int lane  = threadIdx.x & 31;
  const int col   = lane & 15;
  const int rbase = (lane >> 4) << 3;

  __shared__ _Float16 Kl[2][32 * 64];    // s-local x d   (B layout for QK^T)
  __shared__ _Float16 Vl[2][64 * 32];    // d x s-local   (B layout for P*V)
  __shared__ _Float16 Pl[4][16 * 40];    // per-wave P tile, padded rows

  const _Float16* qb = q  + ((size_t)bh * T_ + t0) * HS_;
  const _Float16* kb = kk + (size_t)bh * T_ * HS_;
  const _Float16* vb = vT + (size_t)bh * HS_ * T_;

  // Q tile 16x64: two A fragments, reused for all iterations
  v16h aq0 = load_a16(qb + 0,  HS_);
  v16h aq1 = load_a16(qb + 32, HS_);

  v8f acc[4] = {};
  float mrow[8], lrow[8];
#pragma unroll
  for (int m = 0; m < 8; ++m) { mrow[m] = -INFINITY; lrow[m] = 0.0f; }

  const float SCALE = 0.03125f;          // C**-0.5, C = d_model = 1024
  const int NCH = T_ / 32;               // 64 chunks of 32 columns

  // ---- cooperative async stage of one (K,V) chunk into LDS buffer nb ----
  auto stage = [&](int nb, int s0) {
    const int t = threadIdx.x;           // 0..127
    // K chunk: rows s0..s0+31, 64 halves each -> contiguous 4 KB
    const char* kg = (const char*)(kb + (size_t)s0 * HS_);
    unsigned klo = (unsigned)(uintptr_t)&Kl[nb][0];
    async_cp16(klo + t * 16, kg + t * 16);
    async_cp16(klo + 2048 + t * 16, kg + 2048 + t * 16);
    // V chunk: 64 rows (d), 32 halves each (64 B) at row stride T_
    const char* vg = (const char*)(vb + (size_t)s0);
    unsigned vlo = (unsigned)(uintptr_t)&Vl[nb][0];
    const int row = t >> 2, ch = (t & 3) * 16;
    async_cp16(vlo + row * 64 + ch, vg + (size_t)row * T_ * 2 + ch);
    async_cp16(vlo + (row + 32) * 64 + ch,
               vg + (size_t)(row + 32) * T_ * 2 + ch);
  };

  stage(0, 0);                           // prologue: 4 async issues per wave

  // NOTE: masked positions get 1e-9 (faithful bug) and still feed softmax,
  // so every chunk must be processed -- no early exit.
  for (int it = 0; it < NCH; ++it) {
    const int s0 = it * 32;
    const int cb = it & 1;
    if (it + 1 < NCH) {
      stage(cb ^ 1, s0 + 32);            // prefetch next chunk (+4 in flight)
      asm volatile("s_wait_asynccnt 0x4" ::: "memory");  // current chunk done
    } else {
      asm volatile("s_wait_asynccnt 0x0" ::: "memory");
    }
    __syncthreads();                     // all waves' copies visible

    const _Float16* Kc = &Kl[cb][0];
    const _Float16* Vc = &Vl[cb][0];

    // scores: two 16x16 tiles (cols s0.., s0+16..), K = 64
    v8f c0 = {}, c1 = {};
    c0 = wmma16(aq0, load_b16(Kc, 64), c0);
    c0 = wmma16(aq1, load_b16(Kc + 32, 64), c0);
    c1 = wmma16(aq0, load_b16(Kc + 16 * 64, 64), c1);
    c1 = wmma16(aq1, load_b16(Kc + 16 * 64 + 32, 64), c1);

    float mx[8];
#pragma unroll
    for (int m = 0; m < 8; ++m) {
      const int rg = t0 + rbase + m;
      float a0 = c0[m] * SCALE; if (s0 + col      > rg) a0 = 1e-9f;
      float a1 = c1[m] * SCALE; if (s0 + 16 + col > rg) a1 = 1e-9f;
      c0[m] = a0; c1[m] = a1;
      mx[m] = fmaxf(a0, a1);
    }
#pragma unroll
    for (int m = 0; m < 8; ++m) {
      mx[m] = fmaxf(mx[m], __shfl_xor(mx[m], 1, 32));
      mx[m] = fmaxf(mx[m], __shfl_xor(mx[m], 2, 32));
      mx[m] = fmaxf(mx[m], __shfl_xor(mx[m], 4, 32));
      mx[m] = fmaxf(mx[m], __shfl_xor(mx[m], 8, 32));
    }
    float corr[8], rs[8];
#pragma unroll
    for (int m = 0; m < 8; ++m) {
      const float mn = fmaxf(mrow[m], mx[m]);
      corr[m] = __expf(mrow[m] - mn);
      const float p0 = __expf(c0[m] - mn);
      const float p1 = __expf(c1[m] - mn);
      Pl[w][(rbase + m) * 40 + col]      = (_Float16)p0;
      Pl[w][(rbase + m) * 40 + col + 16] = (_Float16)p1;
      rs[m]   = p0 + p1;
      mrow[m] = mn;
    }
#pragma unroll
    for (int m = 0; m < 8; ++m) {
      rs[m] += __shfl_xor(rs[m], 1, 32);
      rs[m] += __shfl_xor(rs[m], 2, 32);
      rs[m] += __shfl_xor(rs[m], 4, 32);
      rs[m] += __shfl_xor(rs[m], 8, 32);
      lrow[m] = lrow[m] * corr[m] + rs[m];
    }
#pragma unroll
    for (int n = 0; n < 4; ++n)
#pragma unroll
      for (int m = 0; m < 8; ++m) acc[n][m] *= corr[m];

    // P (C-layout) -> A-layout via per-wave LDS tile, then P*V
    v16h pa = load_a16(&Pl[w][0], 40);
#pragma unroll
    for (int n = 0; n < 4; ++n)
      acc[n] = wmma16(pa, load_b16(Vc + n * 16 * 32, 32), acc[n]);

    __syncthreads();                     // done reading buffer cb
  }

  float invl[8];
#pragma unroll
  for (int m = 0; m < 8; ++m) invl[m] = 1.0f / lrow[m];
#pragma unroll
  for (int n = 0; n < 4; ++n)
#pragma unroll
    for (int m = 0; m < 8; ++m) {
      const float o = acc[n][m] * invl[m];
      mha[((size_t)(b * T_ + t0 + rbase + m)) * D_ + h * HS_ + n * 16 + col] =
          (_Float16)o;
    }
}

// ---------------- output projection GEMM (64x64 per wave) ----------------

__global__ __launch_bounds__(256) void out_gemm(const _Float16* __restrict__ mha,
                                                const _Float16* __restrict__ woT,
                                                float* __restrict__ out) {
  const int wid = blockIdx.x * 8 + (threadIdx.x >> 5);
  const int jt = wid & 15;               // 16 tiles of 64 over D
  const int it = wid >> 4;               // 128 tiles of 64 over B*T
  const int i0 = it * 64, j0 = jt * 64;

  const _Float16* ap = mha + (size_t)i0 * D_;
  const _Float16* bp = woT + (size_t)j0 * D_;

  v8f acc[4][4] = {};
  for (int c0 = 0; c0 < D_; c0 += 32) {
    __builtin_prefetch(ap + c0 + 256, 0, 0);
    v16h A[4], Bv[4];
#pragma unroll
    for (int i = 0; i < 4; ++i) A[i]  = load_a16(ap + (size_t)(16 * i) * D_ + c0, D_);
#pragma unroll
    for (int j = 0; j < 4; ++j) Bv[j] = load_b16(bp + (size_t)(16 * j) * D_ + c0, D_);
#pragma unroll
    for (int i = 0; i < 4; ++i)
#pragma unroll
      for (int j = 0; j < 4; ++j) acc[i][j] = wmma16(A[i], Bv[j], acc[i][j]);
  }

  const int lane  = threadIdx.x & 31;
  const int col   = lane & 15;
  const int rbase = (lane >> 4) << 3;
#pragma unroll
  for (int i = 0; i < 4; ++i)
#pragma unroll
    for (int j = 0; j < 4; ++j)
#pragma unroll
      for (int m = 0; m < 8; ++m)
        out[(size_t)(i0 + 16 * i + rbase + m) * D_ + j0 + 16 * j + col] =
            acc[i][j][m];
}

// ---------------- launch ----------------

extern "C" void kernel_launch(void* const* d_in, const int* in_sizes, int n_in,
                              void* d_out, int out_size, void* d_ws, size_t ws_size,
                              hipStream_t stream) {
  const float* x    = (const float*)d_in[0];
  const float* Wqkv = (const float*)d_in[1];
  const float* Wout = (const float*)d_in[2];
  float* out = (float*)d_out;

  char* ws = (char*)d_ws;
  size_t off = 0;
  _Float16* xh  = (_Float16*)(ws + off); off += (size_t)B_ * T_ * D_ * 2;       // 16 MB
  _Float16* wT  = (_Float16*)(ws + off); off += (size_t)H_ * F_ * D_ * 2;       //  6 MB
  _Float16* woT = (_Float16*)(ws + off); off += (size_t)D_ * D_ * 2;            //  2 MB
  _Float16* q   = (_Float16*)(ws + off); off += (size_t)B_ * H_ * T_ * HS_ * 2; // 16 MB
  _Float16* kk  = (_Float16*)(ws + off); off += (size_t)B_ * H_ * T_ * HS_ * 2; // 16 MB
  _Float16* vT  = (_Float16*)(ws + off); off += (size_t)B_ * H_ * T_ * HS_ * 2; // 16 MB
  _Float16* mha = (_Float16*)(ws + off); off += (size_t)B_ * T_ * D_ * 2;       // 16 MB

  cast_f32_to_f16<<<(B_ * T_ * D_) / 256, 256, 0, stream>>>(x, xh, B_ * T_ * D_);
  transpose_wqkv<<<(H_ * D_ * F_ + 255) / 256, 256, 0, stream>>>(Wqkv, wT);
  transpose_wout<<<(D_ * D_) / 256, 256, 0, stream>>>(Wout, woT);

  // 4*16*32*3 = 6144 tiles of 64x64, 8 waves/block
  qkv_gemm<<<(B_ * H_ * 32 * 3) / 8, 256, 0, stream>>>(xh, wT, q, kk, vT);

  // 4 waves/block sharing LDS-staged K/V; grid: (T/64, B*H)
  flash_attn<<<dim3(T_ / 64, B_ * H_), 128, 0, stream>>>(q, kk, vT, mha);

  // (B*T/64)*(D/64) = 2048 tiles of 64x64, 8 waves/block
  out_gemm<<<((B_ * T_ / 64) * (D_ / 64)) / 8, 256, 0, stream>>>(mha, woT, out);
}
